// BiMambaWrapper_34754875359834
// MI455X (gfx1250) — compile-verified, matrix operand order checked by reading, not run on hardware
//
#include <hip/hip_runtime.h>
#include <hip/hip_bf16.h>

// ---------------------------------------------------------------------------
// Bidirectional Mamba block for MI455X (gfx1250, wave32, WMMA + async/TDM).
//   1. cast f32 -> bf16 (hidden, W_in, W_out, W_x_{f,r}, W_dt_{f,r})
//   2. GEMM1 (WMMA + async-to-LDS double buffer): xz = hidden @ W_in^T
//   3. per dir: depthwise causal conv + SiLU -> u (f32 + bf16)
//   4. per dir: GEMM2 (WMMA, guarded sync path): xdbl = u @ W_x^T  (N=96)
//   5. per dir: GEMM3 (WMMA async): delta = softplus(dt @ W_dt^T + b_dt)
//   6. per dir: selective scan; B_t/C_t streamed by the Tensor Data Mover
//      (tensor_load_to_lds, double-buffered 128-step chunks, TENSORcnt)
//   7. combine: ysum_bf16 = bf16(y_f + flip(y_r))
//   8. GEMM4 (WMMA async): out = ysum @ W_out^T -> d_out (f32)
// ---------------------------------------------------------------------------

typedef __bf16 bf16_t;
typedef __attribute__((ext_vector_type(16))) __bf16 v16bf;
typedef __attribute__((ext_vector_type(8)))  __bf16 v8bf;
typedef __attribute__((ext_vector_type(8)))  float   v8f;
typedef __attribute__((ext_vector_type(4)))  unsigned v4u;
typedef __attribute__((ext_vector_type(8)))  unsigned v8u;

#define BATCH   2
#define SEQ_L   2048
#define DMODEL  1024
#define DINNER  2048
#define DSTATE  16
#define DCONV   4
#define DTRANK  64
#define NROWS   (BATCH * SEQ_L)        // 4096 "M" rows for every GEMM

// ---------------------------------------------------------------------------
// CDNA5 async global->LDS 16-byte load (ASYNCcnt path).
// vdst = LDS byte offset (low 32 bits of a generic shared pointer), per lane.
// ---------------------------------------------------------------------------
__device__ __forceinline__ void async_ld16(unsigned lds_off, const void* gptr)
{
    asm volatile("global_load_async_to_lds_b128 %0, %1, off"
                 :: "v"(lds_off), "v"(gptr)
                 : "memory");
}
__device__ __forceinline__ void wait_async_le4()
{
    asm volatile("s_wait_asynccnt 4" ::: "memory");
}
__device__ __forceinline__ void wait_async_0()
{
    asm volatile("s_wait_asynccnt 0" ::: "memory");
}

// ---------------------------------------------------------------------------
// Tensor Data Mover: 2D tile load global->LDS (TENSORcnt path).
// Descriptor per CDNA5 ISA (groups 0/1; VADDR2/3 = NULL).
//   width elems (f32), rows, row stride in elems.
// ---------------------------------------------------------------------------
__device__ __forceinline__ void tdm_load_2d_f32(unsigned lds_off, const void* gaddr,
                                                unsigned width, unsigned rows,
                                                unsigned stride)
{
    const unsigned long long ga = (unsigned long long)gaddr;
    v4u g0;
    g0.x = 1u;                                   // count=1, user mode
    g0.y = lds_off;                              // lds_addr (bytes)
    g0.z = (unsigned)ga;                         // global_addr[31:0]
    g0.w = (unsigned)(ga >> 32) | (2u << 30);    // global_addr[56:32] | type=2
    v8u g1;
    g1.s0 = (2u << 16);                          // workgroup_mask=0, data_size=2 (4B)
    g1.s1 = (width & 0xffffu) << 16;             // atomic_addr=0 | tensor_dim0[15:0]
    g1.s2 = (width >> 16) | ((rows & 0xffffu) << 16);   // dim0 hi | tensor_dim1 lo
    g1.s3 = (rows >> 16) | ((width & 0xffffu) << 16);   // dim1 hi | tile_dim0
    g1.s4 = rows & 0xffffu;                      // tile_dim1 | tile_dim2=0
    g1.s5 = stride;                              // tensor_dim0_stride[31:0]
    g1.s6 = 0u;                                  // dim0_stride hi | dim1_stride lo
    g1.s7 = 0u;                                  // dim1_stride hi
    asm volatile("tensor_load_to_lds %0, %1"
                 :: "s"(g0), "s"(g1)
                 : "memory");
}

// ---------------------------------------------------------------------------
// bf16 WMMA GEMM:  C[M,N] = A[M,K] @ W[N,K]^T   (A,W row-major bf16)
// mode==1: C = softplus(C + bias[n])
// FULL=true: M,N multiples of 128 -> async-to-LDS double-buffered staging.
// FULL=false: guarded synchronous staging (zero-fill rows >= N).
// Block = 256 threads = 8 waves; tile 128x128x32; wave tile 64x32 (4x2 WMMAs).
// ---------------------------------------------------------------------------
template <bool FULL>
__global__ __launch_bounds__(256)
void gemm_bf16_wmma(const bf16_t* __restrict__ A, const bf16_t* __restrict__ Bw,
                    float* __restrict__ C, int M, int N, int K,
                    const float* __restrict__ bias, int mode)
{
    __shared__ __align__(16) bf16_t Alds[FULL ? 2 : 1][128][40]; // 80B rows (16B aligned)
    __shared__ __align__(16) bf16_t Blds[FULL ? 2 : 1][128][40];

    const int tid  = threadIdx.x;
    const int lane = tid & 31;
    const int wave = tid >> 5;
    const int wm   = wave >> 2;            // 0..1  (64-row slab)
    const int wn   = wave & 3;             // 0..3  (32-col slab)
    const int lm   = lane & 15;            // M (A) / N (B) row within tile
    const int half = lane >> 4;            // K-half selector per ISA layout

    const int m_blk = blockIdx.y * 128;
    const int n_blk = blockIdx.x * 128;

    v8f acc[4][2];
    #pragma unroll
    for (int mi = 0; mi < 4; ++mi)
        #pragma unroll
        for (int ni = 0; ni < 2; ++ni) {
            v8f z = {};
            acc[mi][ni] = z;
        }

    const int r  = tid >> 1;               // 0..127: tile row this thread stages
    const int cc = (tid & 1) * 16;         // 0 or 16: K-half of 32

    const int T = K >> 5;                  // K / 32 tiles

    if (FULL) {
        // ---- async double-buffered staging: 4 async b128 per thread/tile ----
        auto stage = [&](int kt, int buf) {
            const bf16_t* pa = A  + (long)(m_blk + r) * K + (kt << 5) + cc;
            const bf16_t* pb = Bw + (long)(n_blk + r) * K + (kt << 5) + cc;
            async_ld16((unsigned)(size_t)&Alds[buf][r][cc],     pa);
            async_ld16((unsigned)(size_t)&Alds[buf][r][cc + 8], pa + 8);
            async_ld16((unsigned)(size_t)&Blds[buf][r][cc],     pb);
            async_ld16((unsigned)(size_t)&Blds[buf][r][cc + 8], pb + 8);
        };
        stage(0, 0);
        for (int i = 0; i < T; ++i) {
            if (i + 1 < T) { stage(i + 1, (i + 1) & 1); wait_async_le4(); }
            else           { wait_async_0(); }
            __syncthreads();

            const int cur = i & 1;
            union Frag { v16bf v; v8bf h[2]; };
            Frag afrag[4], bfrag[2];
            #pragma unroll
            for (int mi = 0; mi < 4; ++mi) {
                const int row = wm * 64 + mi * 16 + lm;
                afrag[mi].h[0] = *(const v8bf*)&Alds[cur][row][8 * half];
                afrag[mi].h[1] = *(const v8bf*)&Alds[cur][row][16 + 8 * half];
            }
            #pragma unroll
            for (int ni = 0; ni < 2; ++ni) {
                const int row = wn * 32 + ni * 16 + lm;
                bfrag[ni].h[0] = *(const v8bf*)&Blds[cur][row][8 * half];
                bfrag[ni].h[1] = *(const v8bf*)&Blds[cur][row][16 + 8 * half];
            }
            #pragma unroll
            for (int mi = 0; mi < 4; ++mi)
                #pragma unroll
                for (int ni = 0; ni < 2; ++ni)
                    acc[mi][ni] = __builtin_amdgcn_wmma_f32_16x16x32_bf16(
                        false, afrag[mi].v, false, bfrag[ni].v,
                        (short)0, acc[mi][ni], false, false);
            __syncthreads();
        }
    } else {
        // ---- guarded synchronous staging (used for the N=96 GEMM) ----
        for (int k0 = 0; k0 < K; k0 += 32) {
            {
                const int gm = m_blk + r;
                v8bf lo = {}, hi = {};
                if (gm < M) {
                    const bf16_t* p = A + (long)gm * K + k0 + cc;
                    lo = *(const v8bf*)p;
                    hi = *(const v8bf*)(p + 8);
                    if (k0 + 32 < K) __builtin_prefetch(p + 32, 0, 1);
                }
                *(v8bf*)&Alds[0][r][cc]     = lo;
                *(v8bf*)&Alds[0][r][cc + 8] = hi;
            }
            {
                const int gn = n_blk + r;
                v8bf lo = {}, hi = {};
                if (gn < N) {
                    const bf16_t* p = Bw + (long)gn * K + k0 + cc;
                    lo = *(const v8bf*)p;
                    hi = *(const v8bf*)(p + 8);
                    if (k0 + 32 < K) __builtin_prefetch(p + 32, 0, 1);
                }
                *(v8bf*)&Blds[0][r][cc]     = lo;
                *(v8bf*)&Blds[0][r][cc + 8] = hi;
            }
            __syncthreads();

            union Frag { v16bf v; v8bf h[2]; };
            Frag afrag[4], bfrag[2];
            #pragma unroll
            for (int mi = 0; mi < 4; ++mi) {
                const int row = wm * 64 + mi * 16 + lm;
                afrag[mi].h[0] = *(const v8bf*)&Alds[0][row][8 * half];
                afrag[mi].h[1] = *(const v8bf*)&Alds[0][row][16 + 8 * half];
            }
            #pragma unroll
            for (int ni = 0; ni < 2; ++ni) {
                const int row = wn * 32 + ni * 16 + lm;
                bfrag[ni].h[0] = *(const v8bf*)&Blds[0][row][8 * half];
                bfrag[ni].h[1] = *(const v8bf*)&Blds[0][row][16 + 8 * half];
            }
            #pragma unroll
            for (int mi = 0; mi < 4; ++mi)
                #pragma unroll
                for (int ni = 0; ni < 2; ++ni)
                    acc[mi][ni] = __builtin_amdgcn_wmma_f32_16x16x32_bf16(
                        false, afrag[mi].v, false, bfrag[ni].v,
                        (short)0, acc[mi][ni], false, false);
            __syncthreads();
        }
    }

    // ---- epilogue: C/D layout: m = vgpr + 8*half, n = lane%16 ----
    #pragma unroll
    for (int mi = 0; mi < 4; ++mi) {
        #pragma unroll
        for (int ni = 0; ni < 2; ++ni) {
            const int n = n_blk + wn * 32 + ni * 16 + lm;
            #pragma unroll
            for (int rr = 0; rr < 8; ++rr) {
                const int m = m_blk + wm * 64 + mi * 16 + rr + 8 * half;
                if (m < M && n < N) {
                    float v = acc[mi][ni][rr];
                    if (mode == 1) {
                        v += bias[n];
                        v = (v > 20.f) ? v : log1pf(__expf(v));   // softplus
                    }
                    C[(long)m * N + n] = v;
                }
            }
        }
    }
}

// ---------------------------------------------------------------------------
__global__ void cast_f32_bf16(const float* __restrict__ in, bf16_t* __restrict__ out, long n)
{
    const long i = (long)blockIdx.x * blockDim.x + threadIdx.x;
    if (i < n) out[i] = (bf16_t)in[i];
}

// ---------------------------------------------------------------------------
// depthwise causal conv (d_conv=4) + SiLU. dir=1 reads x flipped along L.
// ---------------------------------------------------------------------------
__global__ __launch_bounds__(256)
void conv_silu(const float* __restrict__ xz, const float* __restrict__ w,
               const float* __restrict__ bvec, float* __restrict__ uf,
               bf16_t* __restrict__ ub, int dir)
{
    const long i = (long)blockIdx.x * 256 + threadIdx.x;
    const long total = (long)NROWS * DINNER;
    if (i >= total) return;
    const int d = (int)(i & (DINNER - 1));
    const int l = (int)((i >> 11) & (SEQ_L - 1));
    const int b = (int)(i >> 22);

    float acc = bvec[d];
    #pragma unroll
    for (int j = 0; j < DCONV; ++j) {
        const int t = l - (DCONV - 1) + j;
        if (t >= 0) {
            const int sl = dir ? (SEQ_L - 1 - t) : t;
            acc += w[d * DCONV + j] * xz[((long)(b * SEQ_L + sl)) * (2 * DINNER) + d];
        }
    }
    const float u = acc / (1.f + __expf(-acc));   // silu
    uf[i] = u;
    ub[i] = (bf16_t)u;
}

// ---------------------------------------------------------------------------
__global__ void extract_dt_bf16(const float* __restrict__ xdbl, bf16_t* __restrict__ dt, int rows)
{
    const long i = (long)blockIdx.x * 256 + threadIdx.x;
    if (i < (long)rows * DTRANK) {
        const long row = i >> 6;        // DTRANK = 64
        const int  c   = (int)(i & 63);
        dt[i] = (bf16_t)xdbl[row * (DTRANK + 2 * DSTATE) + c];
    }
}

// ---------------------------------------------------------------------------
// selective scan: one thread owns one (b,d) channel, 16 f32 states in regs.
// B_t|C_t (32 contiguous f32 at col 64 of the 96-stride xdbl matrix) are
// streamed by the Tensor Data Mover in 128-timestep chunks, double-buffered.
// Wave 0 issues the TDM op via a wave-level branch (TDM ignores EXEC).
// ---------------------------------------------------------------------------
__global__ __launch_bounds__(256)
void scan_kernel(const float* __restrict__ delta, const float* __restrict__ u,
                 const float* __restrict__ xdbl, const float* __restrict__ xz,
                 const float* __restrict__ A_log, const float* __restrict__ Dvec,
                 float* __restrict__ ybuf, int dir)
{
    const int d = blockIdx.x * 256 + threadIdx.x;
    const int b = blockIdx.y;
    constexpr int CHUNK = 128;
    constexpr int NCH   = SEQ_L / CHUNK;   // 16
    __shared__ __align__(16) float BCbuf[2][CHUNK][2 * DSTATE];   // 2 x 16 KB

    float Arow[DSTATE];
    #pragma unroll
    for (int n = 0; n < DSTATE; ++n) Arow[n] = -__expf(A_log[d * DSTATE + n]);
    const float Dp = Dvec[d];

    float s[DSTATE];
    #pragma unroll
    for (int n = 0; n < DSTATE; ++n) s[n] = 0.f;

    const bool w0 = (threadIdx.x < 32);
    auto issue = [&](int ch, int buf) {
        const float* g = xdbl + ((long)b * SEQ_L + (long)ch * CHUNK)
                               * (DTRANK + 2 * DSTATE) + DTRANK;
        tdm_load_2d_f32((unsigned)(size_t)&BCbuf[buf][0][0], g,
                        2 * DSTATE, CHUNK, DTRANK + 2 * DSTATE);
    };
    if (w0) issue(0, 0);

    for (int ch = 0; ch < NCH; ++ch) {
        if (w0) {
            if (ch + 1 < NCH) { issue(ch + 1, (ch + 1) & 1);
                                __builtin_amdgcn_s_wait_tensorcnt(1); }
            else              { __builtin_amdgcn_s_wait_tensorcnt(0); }
        }
        __syncthreads();
        const float (*BC)[2 * DSTATE] = BCbuf[ch & 1];

        for (int tl = 0; tl < CHUNK; ++tl) {
            const int  t   = ch * CHUNK + tl;
            const long row = (long)b * SEQ_L + t;
            const float dl = delta[row * DINNER + d];
            const float uv = u[row * DINNER + d];
            const float du = dl * uv;
            float y = 0.f;
            #pragma unroll
            for (int n = 0; n < DSTATE; ++n) {
                s[n] = s[n] * __expf(dl * Arow[n]) + du * BC[tl][n];
                y += s[n] * BC[tl][DSTATE + n];
            }
            y += uv * Dp;

            const long orow = dir ? ((long)b * SEQ_L + (SEQ_L - 1 - t)) : row;
            const float zv = xz[orow * (2 * DINNER) + DINNER + d];
            y *= zv / (1.f + __expf(-zv));                   // * silu(z)
            ybuf[orow * DINNER + d] = y;
        }
        __syncthreads();
    }
}

// ---------------------------------------------------------------------------
__global__ void combine_cast(const float* __restrict__ a, const float* __restrict__ b,
                             bf16_t* __restrict__ out, long n)
{
    const long i = (long)blockIdx.x * 256 + threadIdx.x;
    if (i < n) out[i] = (bf16_t)(a[i] + b[i]);
}

// ---------------------------------------------------------------------------
extern "C" void kernel_launch(void* const* d_in, const int* in_sizes, int n_in,
                              void* d_out, int out_size, void* d_ws, size_t ws_size,
                              hipStream_t stream)
{
    (void)in_sizes; (void)n_in; (void)out_size; (void)ws_size;

    const float* hidden = (const float*)d_in[0];
    const float* W_in   = (const float*)d_in[1];
    const float* W_out  = (const float*)d_in[2];

    char* wsp = (char*)d_ws;
    auto alloc = [&](size_t bytes) -> void* {
        void* p = (void*)wsp;
        wsp += (bytes + 255) & ~(size_t)255;
        return p;
    };

    const long M = NROWS;   // 4096
    bf16_t* hid_bf   = (bf16_t*)alloc((size_t)M * DMODEL * 2);
    bf16_t* win_bf   = (bf16_t*)alloc((size_t)(2 * DINNER) * DMODEL * 2);
    bf16_t* wout_bf  = (bf16_t*)alloc((size_t)DMODEL * DINNER * 2);
    bf16_t* wx_bf[2], *wdt_bf[2], *ub[2], *dtb[2];
    float  *uf[2], *xdbl[2], *dl[2], *yb[2];
    for (int d = 0; d < 2; ++d) wx_bf[d]  = (bf16_t*)alloc((size_t)(DTRANK + 2 * DSTATE) * DINNER * 2);
    for (int d = 0; d < 2; ++d) wdt_bf[d] = (bf16_t*)alloc((size_t)DINNER * DTRANK * 2);
    float* xz = (float*)alloc((size_t)M * 2 * DINNER * 4);
    for (int d = 0; d < 2; ++d) uf[d]   = (float*)alloc((size_t)M * DINNER * 4);
    for (int d = 0; d < 2; ++d) ub[d]   = (bf16_t*)alloc((size_t)M * DINNER * 2);
    for (int d = 0; d < 2; ++d) xdbl[d] = (float*)alloc((size_t)M * (DTRANK + 2 * DSTATE) * 4);
    for (int d = 0; d < 2; ++d) dtb[d]  = (bf16_t*)alloc((size_t)M * DTRANK * 2);
    for (int d = 0; d < 2; ++d) dl[d]   = (float*)alloc((size_t)M * DINNER * 4);
    for (int d = 0; d < 2; ++d) yb[d]   = (float*)alloc((size_t)M * DINNER * 4);
    bf16_t* ysum = (bf16_t*)alloc((size_t)M * DINNER * 2);

    auto cast = [&](const float* src, bf16_t* dst, long n) {
        cast_f32_bf16<<<dim3((unsigned)((n + 255) / 256)), dim3(256), 0, stream>>>(src, dst, n);
    };
    cast(hidden, hid_bf, M * DMODEL);
    cast(W_in,  win_bf,  (long)(2 * DINNER) * DMODEL);
    cast(W_out, wout_bf, (long)DMODEL * DINNER);
    cast((const float*)d_in[5],  wx_bf[0],  (long)(DTRANK + 2 * DSTATE) * DINNER);
    cast((const float*)d_in[12], wx_bf[1],  (long)(DTRANK + 2 * DSTATE) * DINNER);
    cast((const float*)d_in[6],  wdt_bf[0], (long)DINNER * DTRANK);
    cast((const float*)d_in[13], wdt_bf[1], (long)DINNER * DTRANK);

    // GEMM1: xz = hidden @ W_in^T   (shared by both directions; reverse = flip)
    gemm_bf16_wmma<true><<<dim3(32, 32), dim3(256), 0, stream>>>(
        hid_bf, win_bf, xz, (int)M, 2 * DINNER, DMODEL, (const float*)nullptr, 0);

    for (int dir = 0; dir < 2; ++dir) {
        const float* cw   = (const float*)d_in[dir ? 10 : 3];
        const float* cb   = (const float*)d_in[dir ? 11 : 4];
        const float* bdt  = (const float*)d_in[dir ? 14 : 7];
        const float* alog = (const float*)d_in[dir ? 15 : 8];
        const float* dvec = (const float*)d_in[dir ? 16 : 9];

        conv_silu<<<dim3((unsigned)((M * DINNER + 255) / 256)), dim3(256), 0, stream>>>(
            xz, cw, cb, uf[dir], ub[dir], dir);

        // xdbl = u @ W_x^T   (N = 96 -> guarded sync path)
        gemm_bf16_wmma<false><<<dim3(1, 32), dim3(256), 0, stream>>>(
            ub[dir], wx_bf[dir], xdbl[dir], (int)M, DTRANK + 2 * DSTATE, DINNER,
            (const float*)nullptr, 0);

        extract_dt_bf16<<<dim3((unsigned)((M * DTRANK + 255) / 256)), dim3(256), 0, stream>>>(
            xdbl[dir], dtb[dir], (int)M);

        // delta = softplus(dt @ W_dt^T + b_dt)
        gemm_bf16_wmma<true><<<dim3(16, 32), dim3(256), 0, stream>>>(
            dtb[dir], wdt_bf[dir], dl[dir], (int)M, DINNER, DTRANK, bdt, 1);

        scan_kernel<<<dim3(DINNER / 256, BATCH), dim3(256), 0, stream>>>(
            dl[dir], uf[dir], xdbl[dir], xz, alog, dvec, yb[dir], dir);
    }

    combine_cast<<<dim3((unsigned)((M * DINNER + 255) / 256)), dim3(256), 0, stream>>>(
        yb[0], yb[1], ysum, M * DINNER);

    // out = (y_f + flip(y_r)) @ W_out^T
    gemm_bf16_wmma<true><<<dim3(8, 32), dim3(256), 0, stream>>>(
        ysum, wout_bf, (float*)d_out, (int)M, DMODEL, DINNER, (const float*)nullptr, 0);
}
